// LocalAttention_30614526885891
// MI455X (gfx1250) — matrix operand order repaired and checked by reading, not compiled, not run on hardware
//
#include <hip/hip_runtime.h>

// Problem constants (from reference)
#define B_   8
#define L_   16384
#define H_   128
#define KS_  65
#define PAD_ 32
#define LP_  (L_ + 2 * PAD_)   // 16448, zero-padded sequence length

typedef __attribute__((ext_vector_type(16))) __bf16 v16bf;
typedef __attribute__((ext_vector_type(8)))  __bf16 v8bf;
typedef __attribute__((ext_vector_type(8)))  float  v8f;
typedef __attribute__((ext_vector_type(4)))  int    v4i;

#if defined(__gfx1250__) && __has_builtin(__builtin_amdgcn_global_load_async_to_lds_b128)
#define HAVE_ASYNC_LDS 1
typedef __attribute__((address_space(1))) v4i gv4i;   // global int4
typedef __attribute__((address_space(3))) v4i lv4i;   // LDS int4
#else
#define HAVE_ASYNC_LDS 0
#endif

__device__ __forceinline__ unsigned short f2bf(float f) {
  union { float f; unsigned u; } v; v.f = f;
  unsigned u = v.u + 0x7FFFu + ((v.u >> 16) & 1u);  // round-to-nearest-even
  return (unsigned short)(u >> 16);
}

__device__ __forceinline__ void wait_async0() {
#if __has_builtin(__builtin_amdgcn_s_wait_asynccnt)
  __builtin_amdgcn_s_wait_asynccnt(0);
#else
  asm volatile("s_wait_asynccnt 0" ::: "memory");
#endif
}

// ---------------------------------------------------------------------------
// Kernel 0: Q [B,L,H] f32 -> Qh [B,LP,H] bf16 with zero halo of PAD rows.
// ---------------------------------------------------------------------------
__global__ void k_pad_q(const float* __restrict__ Q, unsigned short* __restrict__ Qh) {
  size_t idx = (size_t)blockIdx.x * 256 + threadIdx.x;
  const size_t total = (size_t)B_ * LP_ * H_;
  if (idx >= total) return;
  int c  = (int)(idx % H_);
  size_t t = idx / H_;
  int lp = (int)(t % LP_);
  int b  = (int)(t / LP_);
  int l  = lp - PAD_;
  unsigned short v = 0;
  if (l >= 0 && l < L_) v = f2bf(Q[((size_t)b * L_ + l) * H_ + c]);
  Qh[idx] = v;
}

// ---------------------------------------------------------------------------
// Kernel 1: W [H,H,KS] f32 -> Wf bf16 in WMMA B-fragment order:
//   Wf[((k*8+ht)*4+cc)*32 + lane][e] , e=0..15
//   B 32x16 layout used here (contiguous k, same as A): lanes0-15 hold K=0..15, lanes16-31 K=16..31,
//   N = lane%16.  K-in-chunk = cc*32 + (lane/16)*16 + e = c ; N -> h.
// ---------------------------------------------------------------------------
__global__ void k_pack_w(const float* __restrict__ W, unsigned short* __restrict__ Wf) {
  int idx = blockIdx.x * 256 + threadIdx.x;
  if (idx >= (KS_ << 14)) return;          // 65*8*4*32*16
  int e    = idx & 15;
  int lane = (idx >> 4) & 31;
  int cc   = (idx >> 9) & 3;
  int ht   = (idx >> 11) & 7;
  int k    = idx >> 14;
  int n    = lane & 15;
  int half = lane >> 4;
  int c    = cc * 32 + half * 16 + e;
  int h    = ht * 16 + n;
  Wf[idx] = f2bf(W[((size_t)h * H_ + c) * KS_ + k]);
}

// ---------------------------------------------------------------------------
// Kernel 2: conv via v_wmma_f32_16x16x32_bf16, gate by K, store S [B,H,L] f32.
// Block = 256 threads (8 waves); wave w owns l = blk*128 + w*16 .. +16, all h.
// Per tap k: 32KB of B-fragments double-buffered in LDS via async global->LDS
// (tap k+1 staged behind tap k's 32 WMMAs, one barrier/tap), then
// 4 c-chunks x 8 h-tiles WMMAs per wave.  2080 WMMAs / wave.
// ---------------------------------------------------------------------------
__launch_bounds__(256)
__global__ void k_conv_wmma(const unsigned short* __restrict__ Qh,
                            const unsigned short* __restrict__ Wf,
                            const float* __restrict__ Kin,
                            float* __restrict__ S) {
  __shared__ uint4 ldsW4[2][2048];                      // 2 x 32 KB (double buffer)
  const int t    = threadIdx.x;
  const int lane = t & 31;
  const int wave = t >> 5;
  const int b    = blockIdx.x >> 7;                     // /128
  const int lblk = blockIdx.x & 127;
  const int lbase = lblk * 128 + wave * 16;
  const int m    = lane & 15;                           // A row / B-C column
  const int halfsel = (lane >> 4) & 1;

  v8f acc[8] = {};                                      // 16l x 128h f32 acc

  const uint4* __restrict__ Wf4 = reinterpret_cast<const uint4*>(Wf);

  // One tap's worth of B fragments: 2048 uint4 = 32 KB; 8 x b128 per thread.
  auto stage = [&](const uint4* __restrict__ src, uint4* dst) {
#if HAVE_ASYNC_LDS
#pragma unroll
    for (int i = 0; i < 8; ++i)
      __builtin_amdgcn_global_load_async_to_lds_b128(
          (gv4i*)(src + t + i * 256), (lv4i*)(dst + t + i * 256), 0, 0);
#else
#pragma unroll
    for (int i = 0; i < 8; ++i)
      dst[t + i * 256] = src[t + i * 256];
#endif
  };

  auto compute_tap = [&](const unsigned short* ldsW, int k) {
    // A fragment row for this lane (padded coords: l + k - 32 + PAD = l + k)
    const unsigned short* arow = Qh + ((size_t)b * LP_ + (lbase + m + k)) * H_;
#pragma unroll
    for (int cc = 0; cc < 4; ++cc) {
      // A 16x32 bf16 layout: lanes<16 take K = cc*32 + {0..15},
      //                      lanes>=16 take K = cc*32 + {16..31}
      const int cbase = cc * 32 + halfsel * 16;
      v8bf a0 = *reinterpret_cast<const v8bf*>(arow + cbase);
      v8bf a1 = *reinterpret_cast<const v8bf*>(arow + cbase + 8);
      v16bf a = __builtin_shufflevector(a0, a1, 0, 1, 2, 3, 4, 5, 6, 7,
                                        8, 9, 10, 11, 12, 13, 14, 15);
#pragma unroll
      for (int ht = 0; ht < 8; ++ht) {
        const int lidx = ((ht * 4 + cc) * 32 + lane) * 16;
        v8bf b0 = *reinterpret_cast<const v8bf*>(ldsW + lidx);
        v8bf b1 = *reinterpret_cast<const v8bf*>(ldsW + lidx + 8);
        v16bf bb = __builtin_shufflevector(b0, b1, 0, 1, 2, 3, 4, 5, 6, 7,
                                           8, 9, 10, 11, 12, 13, 14, 15);
        acc[ht] = __builtin_amdgcn_wmma_f32_16x16x32_bf16(
            false, a, false, bb, (short)0, acc[ht], false, false);
      }
    }
  };

#if HAVE_ASYNC_LDS
  stage(Wf4, ldsW4[0]);                                 // prologue: tap 0
  for (int k = 0; k < KS_; ++k) {
    wait_async0();                                      // my async stores landed
    __syncthreads();                                    // everyone's landed
    if (k + 1 < KS_)                                    // prefetch next tap
      stage(Wf4 + (size_t)(k + 1) * 2048, ldsW4[(k + 1) & 1]);
    compute_tap(reinterpret_cast<const unsigned short*>(ldsW4[k & 1]), k);
  }
#else
  for (int k = 0; k < KS_; ++k) {
    __syncthreads();
    stage(Wf4 + (size_t)k * 2048, ldsW4[0]);
    __syncthreads();
    compute_tap(reinterpret_cast<const unsigned short*>(ldsW4[0]), k);
  }
#endif

  // Epilogue: C/D layout -> VGPR j holds (M = j + 8*half, N = lane%16).
  // scores = conv * K ; store channels-first S[b][h][l] (8 contiguous l).
  const int moff = halfsel * 8;
#pragma unroll
  for (int ht = 0; ht < 8; ++ht) {
    const int h = ht * 16 + m;
    const size_t kbase = ((size_t)b * L_ + lbase + moff) * H_ + h;
    float4 r0, r1;
    r0.x = acc[ht][0] * Kin[kbase + 0 * H_];
    r0.y = acc[ht][1] * Kin[kbase + 1 * H_];
    r0.z = acc[ht][2] * Kin[kbase + 2 * H_];
    r0.w = acc[ht][3] * Kin[kbase + 3 * H_];
    r1.x = acc[ht][4] * Kin[kbase + 4 * H_];
    r1.y = acc[ht][5] * Kin[kbase + 5 * H_];
    r1.z = acc[ht][6] * Kin[kbase + 6 * H_];
    r1.w = acc[ht][7] * Kin[kbase + 7 * H_];
    float* sp = S + ((size_t)(b * H_ + h)) * L_ + lbase + moff;
    *reinterpret_cast<float4*>(sp)     = r0;
    *reinterpret_cast<float4*>(sp + 4) = r1;
  }
}

// ---------------------------------------------------------------------------
// Kernel 3: per-(b,h) row: max and 1/sum(exp).  One block per row.
// ---------------------------------------------------------------------------
__global__ void k_softmax_stats(const float* __restrict__ S,
                                float* __restrict__ mx, float* __restrict__ inv) {
  const int bh = blockIdx.x;
  const float* row = S + (size_t)bh * L_;
  __shared__ float red[256];
  const int t = threadIdx.x;

  float m = -3.402823e38f;
  for (int i = t; i < L_; i += 256) m = fmaxf(m, row[i]);
  red[t] = m;
  __syncthreads();
  for (int s = 128; s > 0; s >>= 1) {
    if (t < s) red[t] = fmaxf(red[t], red[t + s]);
    __syncthreads();
  }
  const float rm = red[0];
  __syncthreads();

  float sum = 0.0f;
  for (int i = t; i < L_; i += 256) sum += __expf(row[i] - rm);
  red[t] = sum;
  __syncthreads();
  for (int s = 128; s > 0; s >>= 1) {
    if (t < s) red[t] += red[t + s];
    __syncthreads();
  }
  if (t == 0) { mx[bh] = rm; inv[bh] = 1.0f / red[0]; }
}

// ---------------------------------------------------------------------------
// Kernel 4: out[b,l,h] = softmax(S)[b,h,l] * V[b,l,h], via LDS transpose.
// Block handles 64 l x 128 h.  All global accesses coalesced.
// ---------------------------------------------------------------------------
__global__ void k_finalize(const float* __restrict__ S, const float* __restrict__ V,
                           const float* __restrict__ mx, const float* __restrict__ inv,
                           float* __restrict__ out) {
  __shared__ float tile[64 * 129];
  const int blocksPerB = L_ / 64;                 // 256
  const int b  = blockIdx.x / blocksPerB;
  const int lt = blockIdx.x % blocksPerB;
  const int lbase = lt * 64;
  const int t = threadIdx.x;

  for (int it = 0; it < 32; ++it) {               // read S rows (l contiguous)
    int idx = it * 256 + t;
    int lo = idx & 63;
    int h  = idx >> 6;
    float s = S[((size_t)(b * H_ + h)) * L_ + lbase + lo];
    tile[lo * 129 + h] = __expf(s - mx[b * H_ + h]) * inv[b * H_ + h];
  }
  __syncthreads();
  for (int it = 0; it < 32; ++it) {               // write out (h contiguous)
    int idx = it * 256 + t;
    int h  = idx & 127;
    int lo = idx >> 7;
    size_t g = ((size_t)b * L_ + lbase + lo) * H_ + h;
    out[g] = tile[lo * 129 + h] * V[g];
  }
}

// ---------------------------------------------------------------------------
extern "C" void kernel_launch(void* const* d_in, const int* in_sizes, int n_in,
                              void* d_out, int out_size, void* d_ws, size_t ws_size,
                              hipStream_t stream) {
  const float* Q = (const float*)d_in[0];
  const float* K = (const float*)d_in[1];
  const float* V = (const float*)d_in[2];
  const float* W = (const float*)d_in[3];

  char* ws = (char*)d_ws;
  size_t off = 0;
  auto alloc = [&](size_t bytes) -> void* {
    void* p = ws + off;
    off = (off + bytes + 255) & ~(size_t)255;
    return p;
  };
  unsigned short* Qh = (unsigned short*)alloc((size_t)B_ * LP_ * H_ * 2);  // ~34 MB
  unsigned short* Wf = (unsigned short*)alloc((size_t)(KS_ << 14) * 2);    // ~2 MB
  float* S   = (float*)alloc((size_t)B_ * H_ * L_ * 4);                    // 64 MB
  float* mx  = (float*)alloc((size_t)B_ * H_ * 4);
  float* inv = (float*)alloc((size_t)B_ * H_ * 4);

  {
    size_t total = (size_t)B_ * LP_ * H_;
    k_pad_q<<<(unsigned)((total + 255) / 256), 256, 0, stream>>>(Q, Qh);
  }
  {
    int total = KS_ << 14;
    k_pack_w<<<(total + 255) / 256, 256, 0, stream>>>(W, Wf);
  }
  k_conv_wmma<<<B_ * (L_ / 128), 256, 0, stream>>>(Qh, Wf, K, S);
  k_softmax_stats<<<B_ * H_, 256, 0, stream>>>(S, mx, inv);
  k_finalize<<<B_ * (L_ / 64), 256, 0, stream>>>(S, V, mx, inv, (float*)d_out);
}
